// MIN_39127152066962
// MI455X (gfx1250) — matrix operand order repaired and checked by reading, not compiled, not run on hardware
//
#include <hip/hip_runtime.h>
#include <hip/hip_bf16.h>
#include <stdint.h>

// ---------------------------------------------------------------------------
// MI455X / gfx1250, wave32. All GEMMs via v_wmma_f32_16x16x32_bf16.
// B=4096, N=8, D=256, FF=2048.  Total ~103 GFLOP, ~190MB traffic -> mildly
// compute-bound => bf16 WMMA everywhere; everything fits in 192MB L2 so GEMM
// fragments are loaded straight from global as contiguous b128 runs.
// ---------------------------------------------------------------------------

typedef __bf16 bf16_t;
typedef __attribute__((ext_vector_type(16))) __bf16 bf16x16;
typedef __attribute__((ext_vector_type(8)))  __bf16 bf16x8;
typedef __attribute__((ext_vector_type(8)))  float  f32x8;

__device__ __forceinline__ bf16_t f2bf(float f) {
    union { float f; uint32_t u; } v; v.f = f;
    uint32_t r = v.u + 0x7FFFu + ((v.u >> 16) & 1u);   // round-to-nearest-even
    union { unsigned short s; bf16_t b; } o; o.s = (unsigned short)(r >> 16);
    return o.b;
}

__device__ __forceinline__ f32x8 zero8() {
    f32x8 v;
#pragma unroll
    for (int i = 0; i < 8; ++i) v[i] = 0.0f;
    return v;
}

__device__ __forceinline__ bf16x16 mk16(bf16x8 lo, bf16x8 hi) {
    bf16x16 r;
#pragma unroll
    for (int i = 0; i < 8; ++i) { r[i] = lo[i]; r[i + 8] = hi[i]; }
    return r;
}

__device__ __forceinline__ float wave_sum(float v) {
#pragma unroll
    for (int o = 16; o > 0; o >>= 1) v += __shfl_xor(v, o, 32);
    return v;
}

// ---------------------------------------------------------------------------
// LayerNorm over last dim (256) : fp32 in -> bf16 out. One wave32 per row,
// 8 rows per 256-thread block. 8 elems/lane, float4 loads, b128 bf16 store.
// ---------------------------------------------------------------------------
__global__ __launch_bounds__(256)
void k_ln_bf16(const float* __restrict__ x, const float* __restrict__ gamma,
               const float* __restrict__ beta, bf16_t* __restrict__ out, int nrows) {
    const int lane = threadIdx.x & 31;
    const int row  = blockIdx.x * 8 + (threadIdx.x >> 5);
    if (row >= nrows) return;
    const float4* xr = (const float4*)(x + (size_t)row * 256);
    float4 v0 = xr[lane * 2 + 0];
    float4 v1 = xr[lane * 2 + 1];
    float s  = v0.x + v0.y + v0.z + v0.w + v1.x + v1.y + v1.z + v1.w;
    float s2 = v0.x*v0.x + v0.y*v0.y + v0.z*v0.z + v0.w*v0.w
             + v1.x*v1.x + v1.y*v1.y + v1.z*v1.z + v1.w*v1.w;
    s  = wave_sum(s);
    s2 = wave_sum(s2);
    const float mean = s * (1.0f / 256.0f);
    const float var  = s2 * (1.0f / 256.0f) - mean * mean;
    const float inv  = rsqrtf(var + 1e-5f);
    float4 g0 = ((const float4*)gamma)[lane * 2 + 0];
    float4 g1 = ((const float4*)gamma)[lane * 2 + 1];
    float4 b0 = ((const float4*)beta )[lane * 2 + 0];
    float4 b1 = ((const float4*)beta )[lane * 2 + 1];
    union { bf16_t h[8]; uint4 v; } o;
    o.h[0] = f2bf((v0.x - mean) * inv * g0.x + b0.x);
    o.h[1] = f2bf((v0.y - mean) * inv * g0.y + b0.y);
    o.h[2] = f2bf((v0.z - mean) * inv * g0.z + b0.z);
    o.h[3] = f2bf((v0.w - mean) * inv * g0.w + b0.w);
    o.h[4] = f2bf((v1.x - mean) * inv * g1.x + b1.x);
    o.h[5] = f2bf((v1.y - mean) * inv * g1.y + b1.y);
    o.h[6] = f2bf((v1.z - mean) * inv * g1.z + b1.z);
    o.h[7] = f2bf((v1.w - mean) * inv * g1.w + b1.w);
    ((uint4*)(out + (size_t)row * 256))[lane] = o.v;
}

// ---------------------------------------------------------------------------
// Materialize the block-circulant matrices, TRANSPOSED (N-major, K contiguous)
// so WMMA B-fragments are contiguous b128 loads:
//   Mt[nn*2048 + kk] = w[d, o, (k-i)&7],  nn = k*256+o, kk = i*256+d
// ---------------------------------------------------------------------------
__global__ __launch_bounds__(256)
void k_build_circ(const float* __restrict__ w_in, const float* __restrict__ w_out,
                  bf16_t* __restrict__ Min, bf16_t* __restrict__ Mout) {
    const int idx = blockIdx.x * 256 + threadIdx.x;      // over 2048*2048
    const int nn = idx >> 11, kk = idx & 2047;
    const int k = nn >> 8, o = nn & 255;
    const int i = kk >> 8, d = kk & 255;
    const int s = (k - i) & 7;
    const size_t widx = (((size_t)d * 256) + o) * 8 + s;
    Min [idx] = f2bf(w_in [widx]);
    Mout[idx] = f2bf(w_out[widx]);
}

__global__ __launch_bounds__(256)
void k_cvt_bf16(const float* __restrict__ in, bf16_t* __restrict__ out, int n) {
    const int i = blockIdx.x * 256 + threadIdx.x;
    if (i < n) out[i] = f2bf(in[i]);
}

// ---------------------------------------------------------------------------
// tt-product GEMM: C(4096x2048) = A(4096x2048, bf16 row-major) * Mt^T, then
// x = ent_seq + C (fp32, to workspace). Block = 128 thr (4 waves), block tile
// 64x64; each wave owns a 16-row strip x 64 cols => 4 f32 accumulators.
// A-fragment: two b128 runs [k0+8g,+8),[k0+16+8g,+8) of row (m=lane%16).
// B-fragment: two b128 runs [k0+8g,+8),[k0+16+8g,+8) of Mt row n  (N-major storage).
// Rows < 2048 use M_in, else M_out (64 | 2048, so never mixed in a block).
// ---------------------------------------------------------------------------
__global__ __launch_bounds__(128)
void k_tt_gemm(const bf16_t* __restrict__ A, const bf16_t* __restrict__ Bin,
               const bf16_t* __restrict__ Bout, const float* __restrict__ ent,
               float* __restrict__ X) {
    const int lane = threadIdx.x & 31;
    const int wv   = threadIdx.x >> 5;        // 0..3
    const int g    = lane >> 4;               // lane group
    const int ln16 = lane & 15;
    const int mStrip = blockIdx.y * 64 + wv * 16;
    const int nBase  = blockIdx.x * 64;
    const bf16_t* __restrict__ Bm = (blockIdx.y < 32) ? Bin : Bout;
    const bf16_t* __restrict__ arow = A + (size_t)(mStrip + ln16) * 2048;

    f32x8 acc[4];
#pragma unroll
    for (int j = 0; j < 4; ++j) acc[j] = zero8();

#pragma unroll 2
    for (int t = 0; t < 64; ++t) {
        const int k0 = t * 32;
        bf16x8 alo = *(const bf16x8*)(arow + k0 + g * 8);
        bf16x8 ahi = *(const bf16x8*)(arow + k0 + 16 + g * 8);
        bf16x16 af = mk16(alo, ahi);
#pragma unroll
        for (int j = 0; j < 4; ++j) {
            const bf16_t* brow = Bm + (size_t)(nBase + j * 16 + ln16) * 2048 + k0 + g * 8;
            bf16x16 bf = mk16(*(const bf16x8*)(brow), *(const bf16x8*)(brow + 16));
            acc[j] = __builtin_amdgcn_wmma_f32_16x16x32_bf16(
                false, af, false, bf, (short)0, acc[j], false, false);
        }
    }
    // epilogue: residual add, fp32 store. C/D layout: VGPR r -> m=g*8+r, n=lane%16
#pragma unroll
    for (int j = 0; j < 4; ++j) {
        const int col = nBase + j * 16 + ln16;
#pragma unroll
        for (int r = 0; r < 8; ++r) {
            const int row = mStrip + g * 8 + r;
            const size_t idx = (size_t)row * 2048 + col;
            X[idx] = ent[idx] + acc[j][r];
        }
    }
}

// ---------------------------------------------------------------------------
// Fused FFN: per block (256 thr = 8 waves) one 16-row tile.
// Phase 1: h1 = relu(yn_tile @ W1^T + b1) -> 16x2048 bf16 in LDS (exactly
//   64 KB). W1 kept in natural (FF,D) layout = N-major => contiguous B frags.
//   LDS stored XOR-swizzled in 8-elem chunks: elem (m,k) at
//   m*2048 + ((k/8 ^ m)&~... )*8  -> phase-2 ds_load_b128 is conflict-free.
// Phase 2: out = h1 @ W2^T + b2 + x  (W2 natural (D,FF) layout = N-major).
// ---------------------------------------------------------------------------
__global__ __launch_bounds__(256)
void k_ffn(const bf16_t* __restrict__ yn, const bf16_t* __restrict__ W1b,
           const bf16_t* __restrict__ W2b, const float* __restrict__ b1,
           const float* __restrict__ b2, const float* __restrict__ X,
           float* __restrict__ out) {
    __shared__ __align__(16) bf16_t h1[16 * 2048];     // 64 KB
    const int lane = threadIdx.x & 31;
    const int wv   = threadIdx.x >> 5;                 // 0..7
    const int g    = lane >> 4;
    const int ln16 = lane & 15;
    const int r0   = blockIdx.x * 16;
    const bf16_t* __restrict__ arow = yn + (size_t)(r0 + ln16) * 256;

    // ---- phase 1: 128 n-tiles of FF, 16 per wave; K = 256 (8 wmma steps)
    for (int ntb = 0; ntb < 16; ++ntb) {
        const int nt = wv + ntb * 8;
        f32x8 acc = zero8();
#pragma unroll
        for (int t = 0; t < 8; ++t) {
            const int k0 = t * 32;
            bf16x16 af = mk16(*(const bf16x8*)(arow + k0 + g * 8),
                              *(const bf16x8*)(arow + k0 + 16 + g * 8));
            const bf16_t* brow = W1b + (size_t)(nt * 16 + ln16) * 256 + k0 + g * 8;
            bf16x16 bf = mk16(*(const bf16x8*)(brow), *(const bf16x8*)(brow + 16));
            acc = __builtin_amdgcn_wmma_f32_16x16x32_bf16(
                false, af, false, bf, (short)0, acc, false, false);
        }
        const int col   = nt * 16 + ln16;
        const float bv  = b1[col];
        const int chunk = col >> 3, off = col & 7;
#pragma unroll
        for (int r = 0; r < 8; ++r) {
            const int m = g * 8 + r;
            float v = acc[r] + bv;
            v = v > 0.0f ? v : 0.0f;
            h1[m * 2048 + (((chunk ^ (m & 15)) << 3) | off)] = f2bf(v);
        }
    }
    __syncthreads();

    // ---- phase 2: 16 n-tiles of D, 2 per wave; K = 2048 (64 wmma steps)
#pragma unroll
    for (int jj = 0; jj < 2; ++jj) {
        const int nt = wv + jj * 8;
        f32x8 acc = zero8();
        const bf16_t* lrow = h1 + ln16 * 2048;
        for (int t = 0; t < 64; ++t) {
            const int k0  = t * 32;
            const int Clo = (k0 >> 3) + g;            // chunk of run [k0+8g, +8)
            const int Chi = Clo + 2;                  // chunk of run [k0+16+8g, +8)
            bf16x16 af = mk16(*(const bf16x8*)(lrow + ((Clo ^ ln16) << 3)),
                              *(const bf16x8*)(lrow + ((Chi ^ ln16) << 3)));
            const bf16_t* brow = W2b + (size_t)(nt * 16 + ln16) * 2048 + k0 + g * 8;
            bf16x16 bf = mk16(*(const bf16x8*)(brow), *(const bf16x8*)(brow + 16));
            acc = __builtin_amdgcn_wmma_f32_16x16x32_bf16(
                false, af, false, bf, (short)0, acc, false, false);
        }
        const int col  = nt * 16 + ln16;
        const float bv = b2[col];
#pragma unroll
        for (int r = 0; r < 8; ++r) {
            const int m = g * 8 + r;
            const size_t idx = (size_t)(r0 + m) * 256 + col;
            out[idx] = X[idx] + bv + acc[r];
        }
    }
}

// ---------------------------------------------------------------------------
extern "C" void kernel_launch(void* const* d_in, const int* in_sizes, int n_in,
                              void* d_out, int out_size, void* d_ws, size_t ws_size,
                              hipStream_t stream) {
    const float* ent   = (const float*)d_in[0];   // (4096, 8, 256)
    const float* w_in  = (const float*)d_in[1];   // (256, 256, 8)
    const float* w_out = (const float*)d_in[2];   // (256, 256, 8)
    const float* W1    = (const float*)d_in[3];   // (2048, 256)
    const float* b1    = (const float*)d_in[4];   // (2048,)
    const float* W2    = (const float*)d_in[5];   // (256, 2048)
    const float* b2    = (const float*)d_in[6];   // (256,)
    const float* g1    = (const float*)d_in[7];
    const float* be1   = (const float*)d_in[8];
    const float* g2    = (const float*)d_in[9];
    const float* be2   = (const float*)d_in[10];

    char* ws = (char*)d_ws;
    size_t off = 0;
    auto alloc = [&](size_t bytes) -> void* {
        void* p = ws + off;
        off = (off + bytes + 255) & ~(size_t)255;
        return p;
    };
    bf16_t* xn   = (bf16_t*)alloc(32768ull * 256 * 2);   // 16 MB
    bf16_t* Min  = (bf16_t*)alloc(2048ull * 2048 * 2);   //  8 MB
    bf16_t* Mout = (bf16_t*)alloc(2048ull * 2048 * 2);   //  8 MB
    bf16_t* W1b  = (bf16_t*)alloc(2048ull * 256 * 2);    //  1 MB
    bf16_t* W2b  = (bf16_t*)alloc(256ull * 2048 * 2);    //  1 MB
    float*  X    = (float*) alloc(32768ull * 256 * 4);   // 32 MB
    bf16_t* yn   = (bf16_t*)alloc(32768ull * 256 * 2);   // 16 MB

    // 1) LN1 -> xn (bf16)
    k_ln_bf16<<<4096, 256, 0, stream>>>(ent, g1, be1, xn, 32768);
    // 2) weight packing (circulant matrices transposed; W1/W2 natural layout)
    k_build_circ<<<(2048 * 2048) / 256, 256, 0, stream>>>(w_in, w_out, Min, Mout);
    k_cvt_bf16<<<(2048 * 256) / 256, 256, 0, stream>>>(W1, W1b, 2048 * 256);
    k_cvt_bf16<<<(256 * 2048) / 256, 256, 0, stream>>>(W2, W2b, 256 * 2048);
    // 3) tt-product GEMM + residual -> X (fp32)
    k_tt_gemm<<<dim3(32, 64), 128, 0, stream>>>(xn, Min, Mout, ent, X);
    // 4) LN2 -> yn (bf16)
    k_ln_bf16<<<4096, 256, 0, stream>>>(X, g2, be2, yn, 32768);
    // 5) fused FFN (+bias, relu, bias, residual) -> d_out (fp32)
    k_ffn<<<2048, 256, 0, stream>>>(yn, W1b, W2b, b1, b2, X, (float*)d_out);
}
